// GainCellAttention_15212774162485
// MI455X (gfx1250) — 16-bit and 8-bit matrix operand order repaired and checked by reading, compiled, not run on hardware
//
#include <hip/hip_runtime.h>
#include <stdint.h>

#define EPSQ 1e-5f
#define HEADS 16
#define DHEAD 128

typedef __attribute__((ext_vector_type(8)))  int      v8i_t;
typedef __attribute__((ext_vector_type(4)))  int      v4i_t;
typedef __attribute__((ext_vector_type(8)))  float    v8f_t;
typedef __attribute__((ext_vector_type(16))) _Float16 v16h_t;

__device__ __forceinline__ float redmax16(float v) {
  v = fmaxf(v, __shfl_xor(v, 1, 32));
  v = fmaxf(v, __shfl_xor(v, 2, 32));
  v = fmaxf(v, __shfl_xor(v, 4, 32));
  v = fmaxf(v, __shfl_xor(v, 8, 32));
  return v;
}
__device__ __forceinline__ float redsum16(float v) {
  v += __shfl_xor(v, 1, 32);
  v += __shfl_xor(v, 2, 32);
  v += __shfl_xor(v, 4, 32);
  v += __shfl_xor(v, 8, 32);
  return v;
}

__device__ __forceinline__ unsigned lds_off_u32(const void* p) {
  return (unsigned)(unsigned long long)p;
}
__device__ __forceinline__ void async_ld_b128(unsigned lds, const void* g) {
  asm volatile("global_load_async_to_lds_b128 %0, %1, off"
               :: "v"(lds), "v"(g) : "memory");
}

__device__ __forceinline__ v8i_t load_a_iu8(const signed char* A, int ld, int lane) {
  int m = lane & 15, hf = lane >> 4;
  const signed char* r = A + (long)m * ld;
  v8i_t a;
#pragma unroll
  for (int g = 0; g < 4; ++g) {
    const int* p = (const int*)(r + (g >> 1) * 32 + hf * 16 + (g & 1) * 8);
    a[2 * g]     = p[0];
    a[2 * g + 1] = p[1];
  }
  return a;
}

__device__ __forceinline__ v8i_t load_b_iu8(const signed char* Wrow, int ld, int lane) {
  int n = lane & 15, hf = lane >> 4;
  const signed char* r = Wrow + (long)n * ld + hf * 16;
  v4i_t x0 = *(const v4i_t*)(r);
  v4i_t x1 = *(const v4i_t*)(r + 32);
  v8i_t b;
  b[0] = x0[0]; b[1] = x0[1]; b[2] = x0[2]; b[3] = x0[3];
  b[4] = x1[0]; b[5] = x1[1]; b[6] = x1[2]; b[7] = x1[3];
  return b;
}

__device__ __forceinline__ v16h_t load_a_f16_lds(const _Float16* P, int stride, int lane) {
  int m = lane & 15, hf = lane >> 4;
  v16h_t a;
#pragma unroll
  for (int d = 0; d < 8; ++d) {
    int k = hf * 16 + d * 2;
    a[2 * d]     = P[m * stride + k];
    a[2 * d + 1] = P[m * stride + k + 1];
  }
  return a;
}

__device__ __forceinline__ v16h_t load_b_f16(const _Float16* col, long ldT, int lane) {
  int n = lane & 15, hf = lane >> 4;
  return *(const v16h_t*)(col + (long)n * ldT + hf * 16);
}

__global__ void k_zero(float* p, int n) {
  int i = blockIdx.x * blockDim.x + threadIdx.x;
  if (i < n) p[i] = 0.f;
}

__global__ void k_abssum(const float* __restrict__ W, float* __restrict__ sum, long n) {
  __shared__ float red[256];
  long i = (long)blockIdx.x * blockDim.x + threadIdx.x;
  long stride = (long)gridDim.x * blockDim.x;
  float s = 0.f;
  for (; i < n; i += stride) s += fabsf(W[i]);
  red[threadIdx.x] = s;
  __syncthreads();
  for (int off = 128; off > 0; off >>= 1) {
    if ((int)threadIdx.x < off) red[threadIdx.x] += red[threadIdx.x + off];
    __syncthreads();
  }
  if (threadIdx.x == 0) atomicAdd(sum, red[0]);
}

__global__ void k_ternarize(const float* __restrict__ W, const float* __restrict__ sum,
                            signed char* __restrict__ Wt, long n, float invN) {
  long i = (long)blockIdx.x * blockDim.x + threadIdx.x;
  if (i >= n) return;
  float s = fmaxf(sum[0] * invN, EPSQ);
  float q = fminf(fmaxf(rintf(W[i] / s), -1.f), 1.f);
  Wt[i] = (signed char)q;
}

__global__ void k_rowquant_i8(const float* __restrict__ x, signed char* __restrict__ q,
                              float* __restrict__ dq, int C) {
  __shared__ float red[256];
  long row = blockIdx.x;
  const float* xr = x + row * (long)C;
  float am = 0.f;
  for (int c = threadIdx.x; c < C; c += blockDim.x) am = fmaxf(am, fabsf(xr[c]));
  red[threadIdx.x] = am;
  __syncthreads();
  for (int off = 128; off > 0; off >>= 1) {
    if ((int)threadIdx.x < off) red[threadIdx.x] = fmaxf(red[threadIdx.x], red[threadIdx.x + off]);
    __syncthreads();
  }
  float mx = fmaxf(red[0], EPSQ);
  float scale = 127.f / mx;
  for (int c = threadIdx.x; c < C; c += blockDim.x) {
    float v = fminf(fmaxf(rintf(xr[c] * scale), -128.f), 127.f);
    q[row * (long)C + c] = (signed char)v;
  }
  if (threadIdx.x == 0) dq[row] = mx / 127.f;
}

__global__ void k_rowquant_out(const float* __restrict__ pre, float* __restrict__ out, int C) {
  __shared__ float red[256];
  long row = blockIdx.x;
  const float* xr = pre + row * (long)C;
  float am = 0.f;
  for (int c = threadIdx.x; c < C; c += blockDim.x) am = fmaxf(am, fabsf(xr[c]));
  red[threadIdx.x] = am;
  __syncthreads();
  for (int off = 128; off > 0; off >>= 1) {
    if ((int)threadIdx.x < off) red[threadIdx.x] = fmaxf(red[threadIdx.x], red[threadIdx.x + off]);
    __syncthreads();
  }
  float mx = fmaxf(red[0], EPSQ);
  float scale = 127.f / mx, inv = mx / 127.f;
  for (int c = threadIdx.x; c < C; c += blockDim.x) {
    float v = fminf(fmaxf(rintf(xr[c] * scale), -128.f), 127.f);
    out[row * (long)C + c] = v * inv;
  }
}

__global__ void __launch_bounds__(256)
k_gemm_qkv(const signed char* __restrict__ Xq, const float* __restrict__ xqs,
           const signed char* __restrict__ Wt, const float* __restrict__ wsum,
           float invN, signed char* __restrict__ out8, float* __restrict__ outs,
           _Float16* __restrict__ outT, int T, int C, int H) {
  __shared__ __align__(16) signed char wbuf[2][128 * 64];
  int wv = threadIdx.x >> 5, lane = threadIdx.x & 31;
  int tid = threadIdx.x;
  int h  = blockIdx.x % H;
  int mt = (blockIdx.x / H) * 8 + wv;
  int hf = lane >> 4, n = lane & 15;

  auto stage = [&](int k, int buf) {
#pragma unroll
    for (int i = 0; i < 2; ++i) {
      int c = tid + 256 * i;
      int row = c >> 2, seg = c & 3;
      const signed char* g = Wt + (long)(h * DHEAD + row) * C + k + seg * 16;
      async_ld_b128(lds_off_u32(&wbuf[buf][row * 64 + seg * 16]), g);
    }
  };

  v8i_t acc[8];
#pragma unroll
  for (int nt = 0; nt < 8; ++nt) { v8i_t z = {0,0,0,0,0,0,0,0}; acc[nt] = z; }

  stage(0, 0);
  asm volatile("s_wait_asynccnt 0x0" ::: "memory");
  __syncthreads();

  for (int k = 0; k < C; k += 64) {
    int cb = (k >> 6) & 1;
    if (k + 64 < C) stage(k + 64, cb ^ 1);

    v8i_t a = load_a_iu8(Xq + (long)mt * 16 * C + k, C, lane);
    v8i_t b[8];
#pragma unroll
    for (int nt = 0; nt < 8; ++nt)
      b[nt] = load_b_iu8(&wbuf[cb][(nt * 16) * 64], 64, lane);
#pragma unroll
    for (int nt = 0; nt < 8; ++nt)
      acc[nt] = __builtin_amdgcn_wmma_i32_16x16x64_iu8(true, a, true, b[nt], acc[nt], false, false);

    asm volatile("s_wait_asynccnt 0x0" ::: "memory");
    __syncthreads();
  }

  float s_w = fmaxf(wsum[0] * invN, EPSQ);
  float xs[8];
#pragma unroll
  for (int j = 0; j < 8; ++j) xs[j] = xqs[mt * 16 + j + 8 * hf] * s_w;

  float am[8];
#pragma unroll
  for (int j = 0; j < 8; ++j) am[j] = 0.f;
#pragma unroll
  for (int nt = 0; nt < 8; ++nt)
#pragma unroll
    for (int j = 0; j < 8; ++j) am[j] = fmaxf(am[j], fabsf((float)acc[nt][j] * xs[j]));
#pragma unroll
  for (int j = 0; j < 8; ++j) am[j] = fmaxf(redmax16(am[j]), EPSQ);

#pragma unroll
  for (int j = 0; j < 8; ++j) {
    int m = mt * 16 + j + 8 * hf;
    int b2 = m / T, t = m - b2 * T;
    long rb = (long)(b2 * H + h) * T + t;
    float inv = 127.f / am[j], dq = am[j] / 127.f;
    if (n == 0) outs[rb] = dq;
#pragma unroll
    for (int nt = 0; nt < 8; ++nt) {
      float val = (float)acc[nt][j] * xs[j];
      float qv  = fminf(fmaxf(rintf(val * inv), -128.f), 127.f);
      out8[rb * DHEAD + nt * 16 + n] = (signed char)qv;
      if (outT) outT[((long)(b2 * H + h) * DHEAD + nt * 16 + n) * T + t] = (_Float16)(qv * dq);
    }
  }
}

__global__ void __launch_bounds__(128)
k_attn(const signed char* __restrict__ q8, const float* __restrict__ qsc,
       const signed char* __restrict__ k8, const float* __restrict__ ksc,
       const _Float16* __restrict__ vT,
       signed char* __restrict__ yq, float* __restrict__ ysc,
       int H, int T) {
  __shared__ __align__(16) signed char kbuf[4][2][32 * DHEAD];
  __shared__ _Float16 pbuf[4][16 * 40];
  int wv = threadIdx.x >> 5, lane = threadIdx.x & 31;
  int tilesPerBH = T / 16;
  int w = blockIdx.x * 4 + wv;
  int bh = w / tilesPerBH, mt = w % tilesPerBH;
  int hf = lane >> 4, n = lane & 15;

  const signed char* Q = q8 + (long)bh * T * DHEAD + (long)mt * 16 * DHEAD;
  const signed char* K = k8 + (long)bh * T * DHEAD;
  const _Float16*    V = vT + (long)bh * DHEAD * T;
  _Float16* P = &pbuf[wv][0];

  auto stageK = [&](int s, int buf) {
    unsigned base = lds_off_u32(&kbuf[wv][buf][0]) + lane * 16;
    const signed char* g = K + (long)s * DHEAD + lane * 16;
#pragma unroll
    for (int i = 0; i < 8; ++i) async_ld_b128(base + i * 512, g + i * 512);
  };

  v8i_t aq0 = load_a_iu8(Q, DHEAD, lane);
  v8i_t aq1 = load_a_iu8(Q + 64, DHEAD, lane);

  const float rsD = 0.08838834764831845f;
  float qsr[8];
#pragma unroll
  for (int j = 0; j < 8; ++j) qsr[j] = qsc[(long)bh * T + mt * 16 + j + 8 * hf] * rsD;

  float mrow[8], lrow[8];
  v8f_t o[8];
#pragma unroll
  for (int j = 0; j < 8; ++j) { mrow[j] = -1e30f; lrow[j] = 0.f; }
#pragma unroll
  for (int nt = 0; nt < 8; ++nt) { v8f_t z = {0.f,0.f,0.f,0.f,0.f,0.f,0.f,0.f}; o[nt] = z; }

  stageK(0, 0);

  for (int s0 = 0; s0 < T; s0 += 32) {
    int cb = (s0 >> 5) & 1;
    if (s0 + 32 < T) {
      stageK(s0 + 32, cb ^ 1);
      asm volatile("s_wait_asynccnt 0x8" ::: "memory");
    } else {
      asm volatile("s_wait_asynccnt 0x0" ::: "memory");
    }
    const signed char* Kl = &kbuf[wv][cb][0];

    v8i_t kb[4];
#pragma unroll
    for (int sub = 0; sub < 2; ++sub) {
      kb[2 * sub]     = load_b_iu8(Kl + (long)(sub * 16) * DHEAD, DHEAD, lane);
      kb[2 * sub + 1] = load_b_iu8(Kl + (long)(sub * 16) * DHEAD + 64, DHEAD, lane);
    }
    v8i_t ci[2];
#pragma unroll
    for (int sub = 0; sub < 2; ++sub) { v8i_t z = {0,0,0,0,0,0,0,0}; ci[sub] = z; }
#pragma unroll
    for (int sub = 0; sub < 2; ++sub) {
      ci[sub] = __builtin_amdgcn_wmma_i32_16x16x64_iu8(true, aq0, true, kb[2 * sub],     ci[sub], false, false);
      ci[sub] = __builtin_amdgcn_wmma_i32_16x16x64_iu8(true, aq1, true, kb[2 * sub + 1], ci[sub], false, false);
    }

    float sc[2][8], tmax[8];
#pragma unroll
    for (int j = 0; j < 8; ++j) tmax[j] = -1e30f;
#pragma unroll
    for (int sub = 0; sub < 2; ++sub) {
      float ks = ksc[(long)bh * T + s0 + sub * 16 + n];
#pragma unroll
      for (int j = 0; j < 8; ++j) {
        float v = (float)ci[sub][j] * qsr[j] * ks;
        sc[sub][j] = v;
        tmax[j] = fmaxf(tmax[j], v);
      }
    }
#pragma unroll
    for (int j = 0; j < 8; ++j) tmax[j] = redmax16(tmax[j]);

    float corr[8];
#pragma unroll
    for (int j = 0; j < 8; ++j) {
      float mnew = fmaxf(mrow[j], tmax[j]);
      corr[j] = __expf(mrow[j] - mnew);
      mrow[j] = mnew;
      lrow[j] *= corr[j];
    }
#pragma unroll
    for (int nt = 0; nt < 8; ++nt)
#pragma unroll
      for (int j = 0; j < 8; ++j) o[nt][j] *= corr[j];

#pragma unroll
    for (int sub = 0; sub < 2; ++sub)
#pragma unroll
      for (int j = 0; j < 8; ++j) {
        float e  = __expf(sc[sub][j] - mrow[j]);
        float pq = fminf(rintf(e * 127.f), 127.f) * (1.f / 127.f);
        P[(j + 8 * hf) * 40 + sub * 16 + n] = (_Float16)pq;
        lrow[j] += redsum16(e);
      }
    asm volatile("s_wait_dscnt 0" ::: "memory");
    v16h_t pa = load_a_f16_lds(P, 40, lane);

    v16h_t vb[8];
#pragma unroll
    for (int nt = 0; nt < 8; ++nt)
      vb[nt] = load_b_f16(V + (long)(nt * 16) * T + s0, T, lane);
#pragma unroll
    for (int nt = 0; nt < 8; ++nt)
      o[nt] = __builtin_amdgcn_wmma_f32_16x16x32_f16(false, pa, false, vb[nt], (short)0, o[nt], false, false);
  }

  float invZ[8], am[8];
#pragma unroll
  for (int j = 0; j < 8; ++j) invZ[j] = lrow[j] > 0.f ? 1.f / lrow[j] : 0.f;
#pragma unroll
  for (int j = 0; j < 8; ++j) am[j] = 0.f;
#pragma unroll
  for (int nt = 0; nt < 8; ++nt)
#pragma unroll
    for (int j = 0; j < 8; ++j) {
      o[nt][j] *= invZ[j];
      am[j] = fmaxf(am[j], fabsf(o[nt][j]));
    }
#pragma unroll
  for (int j = 0; j < 8; ++j) am[j] = fmaxf(redmax16(am[j]), EPSQ);

  int b = bh / H, h = bh % H;
#pragma unroll
  for (int j = 0; j < 8; ++j) {
    int t = mt * 16 + j + 8 * hf;
    float inv = 127.f / am[j], dq = am[j] / 127.f;
    if (n == 0) ysc[((long)b * T + t) * H + h] = dq;
    long rb = ((long)b * T + t) * (H * DHEAD) + h * DHEAD;
#pragma unroll
    for (int nt = 0; nt < 8; ++nt) {
      float qv = fminf(fmaxf(rintf(o[nt][j] * inv), -128.f), 127.f);
      yq[rb + nt * 16 + n] = (signed char)qv;
    }
  }
}

__global__ void __launch_bounds__(256)
k_gemm_proj(const signed char* __restrict__ yq, const float* __restrict__ ysc,
            const signed char* __restrict__ Wt, const float* __restrict__ wsum,
            float invN, float* __restrict__ pre, int C, int H) {
  int wv = threadIdx.x >> 5, lane = threadIdx.x & 31;
  int ntiles = C / 16;
  int w = blockIdx.x * (blockDim.x >> 5) + wv;
  int mt = w / ntiles, nt = w % ntiles;
  int hf = lane >> 4, n = lane & 15;

  float swp = fmaxf(wsum[0] * invN, EPSQ);
  float facc[8] = {0.f,0.f,0.f,0.f,0.f,0.f,0.f,0.f};

  for (int h = 0; h < H; ++h) {
    const signed char* Arow = yq + (long)mt * 16 * C + h * DHEAD;
    const signed char* Brow = Wt + (long)nt * 16 * C + h * DHEAD;
    v8i_t a0 = load_a_iu8(Arow, C, lane);
    v8i_t a1 = load_a_iu8(Arow + 64, C, lane);
    v8i_t b0 = load_b_iu8(Brow, C, lane);
    v8i_t b1 = load_b_iu8(Brow + 64, C, lane);
    v8i_t ic = {0,0,0,0,0,0,0,0};
    ic = __builtin_amdgcn_wmma_i32_16x16x64_iu8(true, a0, true, b0, ic, false, false);
    ic = __builtin_amdgcn_wmma_i32_16x16x64_iu8(true, a1, true, b1, ic, false, false);
#pragma unroll
    for (int j = 0; j < 8; ++j) {
      int m = mt * 16 + j + 8 * hf;
      facc[j] += (float)ic[j] * ysc[(long)m * H + h];
    }
  }
#pragma unroll
  for (int j = 0; j < 8; ++j) {
    int m = mt * 16 + j + 8 * hf;
    pre[(long)m * C + nt * 16 + n] = facc[j] * swp;
  }
}

extern "C" void kernel_launch(void* const* d_in, const int* in_sizes, int n_in,
                              void* d_out, int out_size, void* d_ws, size_t ws_size,
                              hipStream_t stream) {
  const float* x  = (const float*)d_in[0];
  const float* Ws[4] = { (const float*)d_in[1], (const float*)d_in[2],
                         (const float*)d_in[3], (const float*)d_in[4] };
  const int B = 2, T = 2048, C = 2048, H = HEADS;
  const long M = (long)B * T;
  const long WN = (long)C * C;
  const float invWN = 1.f / (float)WN;

  uint8_t* wsb = (uint8_t*)d_ws;
  size_t off = 0;
  auto alloc = [&](size_t bytes) -> void* {
    void* p = wsb + off;
    off = (off + bytes + 255) & ~(size_t)255;
    return p;
  };

  signed char* xq8 = (signed char*)alloc(M * C);
  float*       xqs = (float*)alloc(M * sizeof(float));
  float*       sums = (float*)alloc(4 * sizeof(float));
  signed char* Wt[4];
  for (int i = 0; i < 4; ++i) Wt[i] = (signed char*)alloc(WN);
  signed char* q8  = (signed char*)alloc(M * C);
  float*       qsc = (float*)alloc(M * sizeof(float));
  signed char* k8  = (signed char*)alloc(M * C);
  float*       ksc = (float*)alloc(M * sizeof(float));
  signed char* v8  = (signed char*)alloc(M * C);
  float*       vsc = (float*)alloc(M * sizeof(float));
  _Float16*    vT  = (_Float16*)alloc(M * C * sizeof(_Float16));
  signed char* yq  = (signed char*)alloc(M * C);
  float*       ysc = (float*)alloc(M * H * sizeof(float));
  float*       pre = (float*)alloc(M * C * sizeof(float));

  k_zero<<<1, 32, 0, stream>>>(sums, 4);
  k_rowquant_i8<<<(int)M, 256, 0, stream>>>(x, xq8, xqs, C);
  for (int i = 0; i < 4; ++i)
    k_abssum<<<1024, 256, 0, stream>>>(Ws[i], sums + i, WN);
  for (int i = 0; i < 4; ++i)
    k_ternarize<<<(int)((WN + 255) / 256), 256, 0, stream>>>(Ws[i], sums + i, Wt[i], WN, invWN);

  k_gemm_qkv<<<512, 256, 0, stream>>>(xq8, xqs, Wt[0], sums + 0, invWN, q8, qsc, nullptr, T, C, H);
  k_gemm_qkv<<<512, 256, 0, stream>>>(xq8, xqs, Wt[1], sums + 1, invWN, k8, ksc, nullptr, T, C, H);
  k_gemm_qkv<<<512, 256, 0, stream>>>(xq8, xqs, Wt[2], sums + 2, invWN, v8, vsc, vT, T, C, H);

  k_attn<<<1024, 128, 0, stream>>>(q8, qsc, k8, ksc, vT, yq, ysc, H, T);

  k_gemm_proj<<<4096, 256, 0, stream>>>(yq, ysc, Wt[3], sums + 3, invWN, pre, C, H);

  k_rowquant_out<<<(int)M, 256, 0, stream>>>(pre, (float*)d_out, C);

  (void)in_sizes; (void)n_in; (void)out_size; (void)ws_size; (void)v8; (void)vsc;
}
